// LIFLayerInfUnrolled_27625229648274
// MI455X (gfx1250) — matrix operand order repaired and checked by reading, not compiled, not run on hardware
//
#include <hip/hip_runtime.h>

// LIF layer, fused GEMM + scan for MI455X (gfx1250, wave32).
// B=8192, T=64, I=256, H=256.  z = inputs @ W^T + b  -> LIF scan over T.
// HBM-bound design: 512MB input read once; GEMM runs on v_wmma_f32_16x16x32_bf16
// (f32 accumulate) so matrix throughput sits far above the 23.3 TB/s roofline.

#define B_  8192
#define T_  64
#define I_  256
#define H_  256
#define DECAY  0.9f
#define THRESH 1.0f

typedef __attribute__((ext_vector_type(16))) __bf16    v16bf;
typedef __attribute__((ext_vector_type(2)))  __bf16    v2bf;
typedef __attribute__((ext_vector_type(8)))  float     v8f;
typedef __attribute__((ext_vector_type(8)))  unsigned  v8u;

// f32 -> bf16 pair packed into one dword; native converts so the backend can
// emit v_cvt_pk_bf16_f32 instead of integer RNE emulation.
__device__ __forceinline__ unsigned pkbf(float lo, float hi) {
  v2bf v;
  v[0] = (__bf16)lo;
  v[1] = (__bf16)hi;
  return __builtin_bit_cast(unsigned, v);
}

__global__ __launch_bounds__(512)
void lif_fused_wmma(const float* __restrict__ in,    // [B,T,I]
                    const float* __restrict__ W,     // [H,I]
                    const float* __restrict__ bias,  // [H]
                    float* __restrict__ out) {       // [2,B,H] (spike_sum, h)
  // Double-buffered A tile in WMMA-fragment order:
  // piece(c,l,half) = 16 bytes at ((c*32+l)*2+half)*16
  __shared__ unsigned lds[2][2048];  // 2 x 8KB

  const int tid  = threadIdx.x;          // 0..511
  const int wave = tid >> 5;             // 0..15 -> H tile
  const int lane = tid & 31;
  const int b0   = blockIdx.x * 16;      // batch stripe
  const int n0   = wave * 16;            // H slice
  const int m16  = lane & 15;
  const int lhi  = lane >> 4;            // 0 or 1

  // ---- B fragments: W[n0+ (lane&15)][k] for 16 consecutive k per chunk ----
  // B 32x16 bf16 layout: lane l holds K = 32c + (l<16?0:16) + e, e=0..15, N = l%16.
  v16bf Bfrag[8];
  {
    const float* wp = W + (size_t)(n0 + m16) * I_ + (lhi ? 16 : 0);
#pragma unroll
    for (int c = 0; c < 8; ++c) {
      const float4* p = (const float4*)(wp + 32 * c);
      float4 x0 = p[0], x1 = p[1], x2 = p[2], x3 = p[3];
      v8u u;
      u[0] = pkbf(x0.x, x0.y); u[1] = pkbf(x0.z, x0.w);
      u[2] = pkbf(x1.x, x1.y); u[3] = pkbf(x1.z, x1.w);
      u[4] = pkbf(x2.x, x2.y); u[5] = pkbf(x2.z, x2.w);
      u[6] = pkbf(x3.x, x3.y); u[7] = pkbf(x3.z, x3.w);
      Bfrag[c] = __builtin_bit_cast(v16bf, u);
    }
  }

  const float bias_l = bias[n0 + m16];   // C/D layout: N = lane%16 for all 8 VGPRs

  // ---- A-tile loader mapping: 512 threads <-> 512 16B pieces ----
  // A 16x32 bf16 layout: lane l holds row M=l%16,
  //   K = 32c + (l<16?0:16) + {0..7} (half 0) and +8+{0..7} (half 1).
  const int p_hlf = tid & 1;
  const int p_l   = (tid >> 1) & 31;
  const int p_c   = tid >> 6;
  const int p_m   = p_l & 15;
  const int p_kb  = 32 * p_c + ((p_l & 16) ? 16 : 0) + 8 * p_hlf;
  const int p_off = ((p_c * 32 + p_l) * 2 + p_hlf) * 4;   // u32 index into buffer
  const float* gptr = in + (size_t)(b0 + p_m) * T_ * I_ + p_kb;

  // scan state lives in the WMMA accumulator layout
  v8f hstate = {};
  v8f ss     = {};

  // ---- prologue: stage t=0 tile ----
  {
    float4 a0 = *(const float4*)(gptr);
    float4 a1 = *(const float4*)(gptr + 4);
    uint4 pk;
    pk.x = pkbf(a0.x, a0.y); pk.y = pkbf(a0.z, a0.w);
    pk.z = pkbf(a1.x, a1.y); pk.w = pkbf(a1.z, a1.w);
    *(uint4*)&lds[0][p_off] = pk;
  }
  __syncthreads();

  for (int t = 0; t < T_; ++t) {
    // issue next tile's global loads early; loadcnt wait lands after WMMA chain
    float4 nf0 = {}, nf1 = {};
    if (t + 1 < T_) {
      const float* g = gptr + (size_t)(t + 1) * I_;
      nf0 = *(const float4*)(g);
      nf1 = *(const float4*)(g + 4);
    }

    // z_t = A(t) x B + bias via 8 chained bf16 WMMAs (K = 256)
    const unsigned* buf = &lds[t & 1][0];
    v8f acc = { bias_l, bias_l, bias_l, bias_l, bias_l, bias_l, bias_l, bias_l };
#pragma unroll
    for (int c = 0; c < 8; ++c) {
      const uint4* fp = (const uint4*)(buf + (c * 32 + lane) * 8);
      uint4 lo = fp[0], hi = fp[1];
      v8u av;
      av[0] = lo.x; av[1] = lo.y; av[2] = lo.z; av[3] = lo.w;
      av[4] = hi.x; av[5] = hi.y; av[6] = hi.z; av[7] = hi.w;
      acc = __builtin_amdgcn_wmma_f32_16x16x32_bf16(
          false, __builtin_bit_cast(v16bf, av),
          false, Bfrag[c],
          (short)0, acc, false, false);
    }

    // LIF update, elementwise on accumulator registers
#pragma unroll
    for (int e = 0; e < 8; ++e) {
      float hv = DECAY * hstate[e] + acc[e];
      float sp = (hv >= THRESH) ? 1.0f : 0.0f;
      ss[e] += sp;
      hstate[e] = (hv >= THRESH) ? 0.0f : hv;
    }

    // convert + stage next tile into the other LDS buffer
    if (t + 1 < T_) {
      uint4 pk;
      pk.x = pkbf(nf0.x, nf0.y); pk.y = pkbf(nf0.z, nf0.w);
      pk.z = pkbf(nf1.x, nf1.y); pk.w = pkbf(nf1.z, nf1.w);
      *(uint4*)&lds[(t + 1) & 1][p_off] = pk;
    }
    __syncthreads();
  }

  // ---- write outputs straight from D layout: M = e + 8*(lane>=16), N = lane%16
  const int col  = n0 + m16;
  const int brow = b0 + (lhi << 3);
#pragma unroll
  for (int e = 0; e < 8; ++e) {
    const size_t b = (size_t)(brow + e);
    out[b * H_ + col]                   = ss[e];      // spike_sum
    out[(size_t)B_ * H_ + b * H_ + col] = hstate[e];  // final h
  }
}

extern "C" void kernel_launch(void* const* d_in, const int* in_sizes, int n_in,
                              void* d_out, int out_size, void* d_ws, size_t ws_size,
                              hipStream_t stream) {
  const float* in  = (const float*)d_in[0];
  const float* W   = (const float*)d_in[1];
  const float* b   = (const float*)d_in[2];
  float* out       = (float*)d_out;

  dim3 grid(B_ / 16);   // 512 workgroups, one 16-batch stripe each
  dim3 block(512);      // 16 waves: one per 16-wide H slice
  hipLaunchKernelGGL(lif_fused_wmma, grid, block, 0, stream, in, W, b, out);
}
